// DRNNclassification_17755394802204
// MI455X (gfx1250) — matrix operand order repaired and checked by reading, not compiled, not run on hardware
//
#include <hip/hip_runtime.h>
#include <stdint.h>

// ---------------------------------------------------------------------------
// CDNA5 (gfx1250) dilated-GRU classifier.
// All GEMMs (input-gate, recurrent, dense head) via v_wmma_f32_16x16x32_bf16.
// Transposed formulation: D[gate,seq] = A(W^T/U^T)[16x32] * B(x^T/h^T)[32x16].
// Hidden padded 50->64; gates at col g*64+j so z/r/h~ align lane-for-lane.
// Recurrent state lives in registers; C->B relayout = one half-wave shfl_xor
// of packed bf16 pairs. Weight fragments staged in LDS once per block so the
// scan inner loop is ds_load_b128 @ immediate offsets + WMMA only.
// Biases folded into padded K-rows 50/51 of U (exact: b == 0 in this model).
// ---------------------------------------------------------------------------

typedef __attribute__((ext_vector_type(16))) __bf16 v16bf;
typedef __attribute__((ext_vector_type(8)))  float  v8f;

__device__ inline v8f wmma_bf16(v16bf a, v16bf b, v8f c) {
  // (neg_a, A, neg_b, B, c_mod, C, reuse_a, reuse_b)
  return __builtin_amdgcn_wmma_f32_16x16x32_bf16(false, a, false, b, (short)0, c,
                                                 false, false);
}
__device__ inline float sigmf_(float x) { return 1.0f / (1.0f + __expf(-x)); }
__device__ inline float tanhf_(float x) { return 1.0f - 2.0f / (__expf(2.0f * x) + 1.0f); }
__device__ inline v8f zero8() {
  v8f z;
#pragma unroll
  for (int r = 0; r < 8; ++r) z[r] = 0.0f;
  return z;
}
__device__ inline unsigned short bf16bits(float x) {
  union { __bf16 b; unsigned short u; } c;
  c.b = (__bf16)x;
  return c.u;
}

// ------------------------- weight packing kernels --------------------------
// A-fragment (16x32 bf16) lane map: m = mt*16 + (lane&15);
//   k = kc*32 + (lane>=16 ? 16 : 0) + e
// B-fragment (32x16 bf16) lane map: n = jt*16 + (lane&15);
//   k = kc*32 + (lane>=16 ? 16 : 0) + e

__global__ void pack_w0_kernel(const float* __restrict__ W0, __bf16* __restrict__ frag) {
  int idx = blockIdx.x * 256 + threadIdx.x;           // 4*12*32*16 = 24576
  if (idx >= 24576) return;
  int e = idx & 15, t = idx >> 4;
  int lane = t & 31; t >>= 5;
  int mt = t % 12, kc = t / 12;                       // kc 0..3 (K = 128)
  int m = mt * 16 + (lane & 15);
  int k = kc * 32 + ((lane >> 4) ? 16 : 0) + e;
  int g = m / 64, jh = m % 64;
  float v = (jh < 50 && k < 128) ? W0[k * 150 + g * 50 + jh] : 0.0f;
  frag[idx] = (__bf16)v;
}

__global__ void pack_wn_kernel(const float* __restrict__ Wn, __bf16* __restrict__ frag) {
  int idx = blockIdx.x * 256 + threadIdx.x;           // 5 * 2*12*32*16 = 61440
  if (idx >= 61440) return;
  int li = idx / 12288, rem = idx % 12288;
  int e = rem & 15, t = rem >> 4;
  int lane = t & 31; t >>= 5;
  int mt = t % 12, kc = t / 12;                       // kc 0..1 (K = 64)
  int m = mt * 16 + (lane & 15);
  int k = kc * 32 + ((lane >> 4) ? 16 : 0) + e;
  int g = m / 64, jh = m % 64;
  float v = (jh < 50 && k < 50) ? Wn[(li * 50 + k) * 150 + g * 50 + jh] : 0.0f;
  frag[idx] = (__bf16)v;
}

__global__ void pack_u_kernel(const float* __restrict__ U0, const float* __restrict__ b0,
                              const float* __restrict__ Un, const float* __restrict__ bn,
                              __bf16* __restrict__ frag) {
  int idx = blockIdx.x * 256 + threadIdx.x;           // 6 * 12288 = 73728
  if (idx >= 73728) return;
  int li = idx / 12288, rem = idx % 12288;
  int e = rem & 15, t = rem >> 4;
  int lane = t & 31; t >>= 5;
  int mt = t % 12, kc = t / 12;
  int m = mt * 16 + (lane & 15);
  int k = kc * 32 + ((lane >> 4) ? 16 : 0) + e;
  int g = m / 64, jh = m % 64;
  const float* U  = (li == 0) ? U0 : (Un + (li - 1) * 50 * 150);
  const float* bb = (li == 0) ? b0 : (bn + (li - 1) * 2 * 150);
  float v = 0.0f;
  if (jh < 50) {
    int col = g * 50 + jh;
    if (k < 50)       v = U[k * 150 + col];
    else if (k == 50) v = bb[150 + col];   // recurrent bias b_r (K-row trick)
    else if (k == 51) v = bb[col];         // input bias b_i (exact since b==0)
  }
  frag[idx] = (__bf16)v;
}

__global__ void pack_d2_kernel(const float* __restrict__ W2, __bf16* __restrict__ frag) {
  int idx = blockIdx.x * 256 + threadIdx.x;           // 50*100*32*16 = 2,560,000
  if (idx >= 2560000) return;
  int e = idx & 15, t = idx >> 4;
  int lane = t & 31; t >>= 5;
  int jt = t % 100, kc = t / 100;
  int n = jt * 16 + (lane & 15);
  int k = kc * 32 + ((lane >> 4) ? 16 : 0) + e;       // B-fragment layout
  frag[idx] = (__bf16)W2[k * 1600 + n];
}

// ------------------------------ GRU scan -----------------------------------
// One wave owns 16 sequences for the whole dilated scan (all dn steps).
// Block = 128 threads = 4 waves (fills the 4 SIMD32s of a WGP).
// W/U fragments live in LDS: inner loop is ds_load_b128 + v_wmma only.
template <bool L0>
__global__ __launch_bounds__(128) void gru_scan_kernel(
    const float* __restrict__ x0,        // layer-0 input [64,2048,128] f32
    const __bf16* __restrict__ actIn,    // prev layer activations (bf16, Hp=64)
    __bf16* __restrict__ actOut,
    const v16bf* __restrict__ wfrag,     // [KC][12][32] A-fragments of W^T
    const v16bf* __restrict__ ufrag,     // [2][12][32]  A-fragments of U^T
    int S, int dn) {
  constexpr int KC = L0 ? 4 : 2;
  constexpr int WB = KC * 12 * 32 * 32;                 // 49152 / 24576 bytes
  constexpr int UB = 2 * 12 * 32 * 32;                  // 24576 bytes
  __shared__ __align__(32) char smem[WB + UB];

  // Cooperative stage of weight fragments into LDS (amortized over dn steps).
  {
    const uint4* gw = (const uint4*)wfrag;
    uint4* sw = (uint4*)smem;
    for (int i = threadIdx.x; i < WB / 16; i += 128) sw[i] = gw[i];
    const uint4* gu = (const uint4*)ufrag;
    uint4* su = (uint4*)(smem + WB);
    for (int i = threadIdx.x; i < UB / 16; i += 128) su[i] = gu[i];
  }
  __syncthreads();

  const int lane = threadIdx.x & 31;
  const int wid  = threadIdx.x >> 5;
  const int seqBase = (blockIdx.x * 4 + wid) * 16;
  const int nloc = lane & 15;
  const int hi   = lane >> 4;            // half-wave index
  const int seq  = seqBase + nloc;       // this lane's column (sequence)
  const int rphase = seq >> 6;           // j = r*64 + b
  const int bidx   = seq & 63;

  const v16bf* swf = (const v16bf*)smem + lane;          // + (kc*12+mt)*32
  const v16bf* suf = (const v16bf*)(smem + WB) + lane;

  // recurrent state h^T in C-layout tiles: tile t holds hid rows 16t..16t+15
  v8f hT[4];
#pragma unroll
  for (int t = 0; t < 4; ++t)
#pragma unroll
    for (int r = 0; r < 8; ++r) hT[t][r] = 0.0f;

  // strength-reduced streaming pointers
  const float*  xrow = L0 ? (x0 + ((long long)bidx * 2048 + rphase) * 128) : nullptr;
  const __bf16* arow = L0 ? nullptr : (actIn + (long long)seq * 64);
  __bf16* orow = actOut + ((long long)rphase * dn * 64 + bidx) * 64;
  const long long aStep = (long long)S * 64;             // activations: next i
  const long long oStep = 64LL * 64;                     // output: next i

  for (int i = 0; i < dn; ++i) {
    // ---- x^T B-fragments: lane holds 16 contiguous k for its sequence ----
    v16bf xb[KC];
    if constexpr (L0) {
#pragma unroll
      for (int kc = 0; kc < KC; ++kc) {
        const float4* p = (const float4*)(xrow + kc * 32 + hi * 16);
#pragma unroll
        for (int q = 0; q < 4; ++q) {
          float4 f = p[q];
          xb[kc][q * 4 + 0] = (__bf16)f.x;
          xb[kc][q * 4 + 1] = (__bf16)f.y;
          xb[kc][q * 4 + 2] = (__bf16)f.z;
          xb[kc][q * 4 + 3] = (__bf16)f.w;
        }
      }
      xrow += 32 * 128;                                  // t_orig advances by rate
    } else {
#pragma unroll
      for (int kc = 0; kc < KC; ++kc) {
        union { uint4 u[2]; v16bf v; } tt;
        tt.u[0] = *(const uint4*)(arow + kc * 32 + hi * 16);
        tt.u[1] = *(const uint4*)(arow + kc * 32 + hi * 16 + 8);
        xb[kc] = tt.v;
      }
      arow += aStep;
    }

    // ---- h^T B-fragments from C-layout state: one packed half-wave swap ----
    v16bf hb[2];
#pragma unroll
    for (int kc = 0; kc < 2; ++kc) {
      v8f T0 = hT[2 * kc], T1 = hT[2 * kc + 1];
      union { unsigned short s[16]; v16bf v; } H;
#pragma unroll
      for (int r = 0; r < 8; ++r) {
        unsigned int packed = (unsigned int)bf16bits(T0[r]) |
                              ((unsigned int)bf16bits(T1[r]) << 16);
        unsigned int swp = (unsigned int)__shfl_xor((int)packed, 16, 32);
        // low (e=r):  k-range low16  -> T0 own on low half, T1 from peer on high
        // high(e=8+r):k-range high16 -> T0 from peer on low half, T1 own on high
        H.s[r]     = hi ? (unsigned short)(swp >> 16) : (unsigned short)(packed & 0xffffu);
        H.s[8 + r] = hi ? (unsigned short)(packed >> 16) : (unsigned short)(swp & 0xffffu);
      }
      hb[kc] = H.v;
    }
    if (hi) {  // constant-1 slots feeding bias K-rows 50/51 of U
      hb[1][2] = (__bf16)1.0f;
      hb[1][3] = (__bf16)1.0f;
    }

    // ---- 4 gate-triples: tiles t (z), t+4 (r), t+8 (h~) share lanes ----
#pragma unroll
    for (int t = 0; t < 4; ++t) {
      v8f axz = zero8(), axr = zero8(), axh = zero8();
      v8f arz = zero8(), arr2 = zero8(), arh = zero8();
#pragma unroll
      for (int kc = 0; kc < KC; ++kc) {
        axz = wmma_bf16(swf[(kc * 12 + t    ) * 32], xb[kc], axz);
        axr = wmma_bf16(swf[(kc * 12 + t + 4) * 32], xb[kc], axr);
        axh = wmma_bf16(swf[(kc * 12 + t + 8) * 32], xb[kc], axh);
      }
#pragma unroll
      for (int kc = 0; kc < 2; ++kc) {
        arz  = wmma_bf16(suf[(kc * 12 + t    ) * 32], hb[kc], arz);
        arr2 = wmma_bf16(suf[(kc * 12 + t + 4) * 32], hb[kc], arr2);
        arh  = wmma_bf16(suf[(kc * 12 + t + 8) * 32], hb[kc], arh);
      }
#pragma unroll
      for (int r = 0; r < 8; ++r) {
        float z  = sigmf_(axz[r] + arz[r]);
        float rg = sigmf_(axr[r] + arr2[r]);
        float hh = tanhf_(axh[r] + rg * arh[r]);
        hT[t][r] = z * hT[t][r] + (1.0f - z) * hh;
      }
    }

    // ---- store y[t'] , t' = r*dn + i (faithful split/concat order) ----
#pragma unroll
    for (int t = 0; t < 4; ++t) {
      union { __bf16 h[8]; uint4 u; } pk;
#pragma unroll
      for (int r = 0; r < 8; ++r) pk.h[r] = (__bf16)hT[t][r];
      *(uint4*)(orow + t * 16 + hi * 8) = pk.u;
    }
    orow += oStep;
  }
}

// ------------------------------ head ---------------------------------------
__global__ void feats_kernel(const __bf16* __restrict__ act5, __bf16* __restrict__ feats) {
  int idx = blockIdx.x * 256 + threadIdx.x;           // 64*1600
  if (idx >= 64 * 1600) return;
  int b = idx / 1600, f = idx % 1600;
  int k = f / 50, h = f % 50;
  feats[idx] = act5[(((long long)(2016 + k)) * 64 + b) * 64 + h];
}

__global__ __launch_bounds__(128) void dense2_kernel(
    const __bf16* __restrict__ feats, const v16bf* __restrict__ w2frag,
    const float* __restrict__ d2b, float* __restrict__ h1) {
  int lane = threadIdx.x & 31, wave = threadIdx.x >> 5;
  int wg = blockIdx.x * 4 + wave;                     // 400 tile-waves
  int mtb = wg & 3, jt = wg >> 2;
  int nloc = lane & 15, hi = lane >> 4;
  float bv = d2b[jt * 16 + nloc];
  v8f acc;
#pragma unroll
  for (int r = 0; r < 8; ++r) acc[r] = bv;
  const __bf16* arow = feats + (mtb * 16 + nloc) * 1600;
  const v16bf* wrow = w2frag + jt * 32 + lane;
  for (int kc = 0; kc < 50; ++kc) {
    int kb = kc * 32 + hi * 16;                       // A-fragment contiguous runs
    union { uint4 u[2]; v16bf v; } A;
    A.u[0] = *(const uint4*)(arow + kb);
    A.u[1] = *(const uint4*)(arow + kb + 8);
    acc = wmma_bf16(A.v, wrow[kc * 100 * 32], acc);
  }
#pragma unroll
  for (int r = 0; r < 8; ++r) {
    float v = acc[r];
    v = v > 0.0f ? v : 0.0f;                          // ReLU
    h1[(mtb * 16 + r + 8 * hi) * 1600 + jt * 16 + nloc] = v;
  }
}

__global__ __launch_bounds__(64) void cls_kernel(
    const float* __restrict__ h1, const float* __restrict__ clsW,
    const float* __restrict__ clsb, float* __restrict__ out) {
  __shared__ float sm[64];
  int b = blockIdx.x, j = threadIdx.x;
  float acc = 0.0f;
  if (j < 41) {
    acc = clsb[j];
    const float* hr = h1 + b * 1600;
    for (int k = 0; k < 1600; ++k) acc += hr[k] * clsW[k * 41 + j];
  }
  sm[j] = (j < 41) ? acc : -3.0e38f;
  __syncthreads();
  for (int s = 32; s > 0; s >>= 1) { if (j < s) sm[j] = fmaxf(sm[j], sm[j + s]); __syncthreads(); }
  float mx = sm[0];
  __syncthreads();
  float e = (j < 41) ? __expf(acc - mx) : 0.0f;
  sm[j] = e;
  __syncthreads();
  for (int s = 32; s > 0; s >>= 1) { if (j < s) sm[j] += sm[j + s]; __syncthreads(); }
  if (j < 41) out[b * 41 + j] = e / sm[0];
}

// ------------------------------ launch -------------------------------------
extern "C" void kernel_launch(void* const* d_in, const int* in_sizes, int n_in,
                              void* d_out, int out_size, void* d_ws, size_t ws_size,
                              hipStream_t stream) {
  (void)in_sizes; (void)n_in; (void)out_size; (void)ws_size;
  const float* x    = (const float*)d_in[0];
  const float* W0   = (const float*)d_in[1];
  const float* U0   = (const float*)d_in[2];
  const float* b0   = (const float*)d_in[3];
  const float* Wn   = (const float*)d_in[4];
  const float* Un   = (const float*)d_in[5];
  const float* bn   = (const float*)d_in[6];
  const float* d2W  = (const float*)d_in[7];
  const float* d2b  = (const float*)d_in[8];
  const float* clsW = (const float*)d_in[9];
  const float* clsb = (const float*)d_in[10];
  float* out = (float*)d_out;

  char* ws = (char*)d_ws;
  size_t off = 0;
  auto alloc = [&](size_t bytes) {
    void* p = ws + off;
    off += (bytes + 255) & ~(size_t)255;
    return p;
  };
  __bf16* actA  = (__bf16*)alloc((size_t)2048 * 64 * 64 * 2);   // 16.8 MB
  __bf16* actB  = (__bf16*)alloc((size_t)2048 * 64 * 64 * 2);   // 16.8 MB
  __bf16* w0f   = (__bf16*)alloc((size_t)24576 * 2);
  __bf16* wnf   = (__bf16*)alloc((size_t)61440 * 2);
  __bf16* uf    = (__bf16*)alloc((size_t)73728 * 2);
  __bf16* d2f   = (__bf16*)alloc((size_t)2560000 * 2);          // 5.1 MB
  __bf16* feats = (__bf16*)alloc((size_t)64 * 1600 * 2);
  float*  h1    = (float*)alloc((size_t)64 * 1600 * 4);

  // Pack weights into WMMA fragment order (cheap; rerun every call).
  pack_w0_kernel<<<96,    256, 0, stream>>>(W0, w0f);
  pack_wn_kernel<<<240,   256, 0, stream>>>(Wn, wnf);
  pack_u_kernel <<<288,   256, 0, stream>>>(U0, b0, Un, bn, uf);
  pack_d2_kernel<<<10000, 256, 0, stream>>>(d2W, d2f);

  // Layer 0: rate 32 -> S=2048 sequences, dn=64 steps.
  gru_scan_kernel<true><<<32, 128, 0, stream>>>(
      x, nullptr, actA, (const v16bf*)w0f, (const v16bf*)uf, 2048, 64);

  // Layers 1..5: rates 64..1024; ping-pong activation buffers.
  int S = 4096, dn = 32;
  for (int li = 1; li <= 5; ++li) {
    const __bf16* inA = (li & 1) ? actA : actB;
    __bf16*       oA  = (li & 1) ? actB : actA;
    gru_scan_kernel<false><<<S / 64, 128, 0, stream>>>(
        nullptr, inA, oA,
        (const v16bf*)(wnf + (size_t)(li - 1) * 12288),
        (const v16bf*)(uf + (size_t)li * 12288), S, dn);
    S *= 2; dn /= 2;
  }

  // Head: last 32 timesteps of layer-5 output (in actB) -> 1600 feats.
  feats_kernel <<<400, 256, 0, stream>>>(actB, feats);
  dense2_kernel<<<100, 128, 0, stream>>>(feats, (const v16bf*)d2f, d2b, h1);
  cls_kernel   <<<64,  64,  0, stream>>>(h1, clsW, clsb, out);
}
